// OrthoLinear_6210522710042
// MI455X (gfx1250) — matrix operand order repaired and checked by reading, not compiled, not run on hardware
//
#include <hip/hip_runtime.h>
#include <hip/hip_bf16.h>

typedef __attribute__((ext_vector_type(16))) _Float16 v16h;
typedef __attribute__((ext_vector_type(8)))  _Float16 v8h;
typedef __attribute__((ext_vector_type(8)))  float    v8f;

static constexpr int TOKENS = 4096;
static constexpr int IN_F   = 4096;
static constexpr int OUT_F  = 4096;
static constexpr float ALPHA = 1.0f;

static constexpr int BM    = 128;  // block tile M (8 waves x 16)
static constexpr int BN    = 64;   // block tile N (4 wmma tiles)
static constexpr int KSTEP = 64;   // k per stage (2 wmma k-subs)
static constexpr int LDA   = 72;   // padded LDS row (f16) -> conflict-free b128 reads

// ---------------------------------------------------------------------------
// Stage 1: dequantize packed int4 (low nibble -> even col, high -> odd col)
// ---------------------------------------------------------------------------
__global__ void ortho_dequant(const int* __restrict__ packed,
                              const float* __restrict__ scales,
                              float* __restrict__ w) {
  int idx = blockIdx.x * blockDim.x + threadIdx.x;
  if (idx >= OUT_F * (IN_F / 2)) return;
  int o = idx >> 11;          // / (IN_F/2) == 2048
  int h = idx & 2047;
  int b = packed[idx] & 0xFF;
  float s = scales[o];
  float lo = (float)((b & 0xF) - 8) * s;
  float hi = (float)(((b >> 4) & 0xF) - 8) * s;
  float* row = w + (size_t)o * IN_F;
  row[2 * h]     = lo;
  row[2 * h + 1] = hi;
}

// ---------------------------------------------------------------------------
// Stage 2: sparse COO residual scatter (atomic: duplicate/order safe)
// ---------------------------------------------------------------------------
__global__ void ortho_scatter(const float* __restrict__ vals,
                              const int* __restrict__ rows,
                              const int* __restrict__ cols,
                              float* __restrict__ w, int nnz) {
  int i = blockIdx.x * blockDim.x + threadIdx.x;
  if (i >= nnz) return;
  atomicAdd(&w[(size_t)rows[i] * IN_F + cols[i]], ALPHA * vals[i]);
}

// ---------------------------------------------------------------------------
// Stage 3: f32 -> (f16 hi, f16 lo) split for near-f32-accurate f16 WMMA
// ---------------------------------------------------------------------------
__global__ void ortho_split(const float* __restrict__ src,
                            _Float16* __restrict__ hi,
                            _Float16* __restrict__ lo, int n) {
  int i = blockIdx.x * blockDim.x + threadIdx.x;
  if (i >= n) return;
  float v = src[i];
  _Float16 h = (_Float16)v;
  hi[i] = h;
  lo[i] = (_Float16)(v - (float)h);
}

// ---------------------------------------------------------------------------
// Async global -> LDS copy of 16 bytes (8 f16) per lane. ASYNCcnt-tracked.
// ---------------------------------------------------------------------------
static __device__ inline void cp16_async(const _Float16* g, const _Float16* l) {
  asm volatile("global_load_async_to_lds_b128 %0, %1, off"
               :: "v"((unsigned)(unsigned long long)l), "v"(g)
               : "memory");
}
static __device__ inline void wait_async_all() {
  asm volatile("s_wait_asynccnt 0x0" ::: "memory");
}

static __device__ inline v16h combine16(v8h x0, v8h x1) {
  v16h r;
#pragma unroll
  for (int i = 0; i < 8; ++i) { r[i] = x0[i]; r[i + 8] = x1[i]; }
  return r;
}

// A fragment (16x32 f16): lane -> row m = lane&15, g = lane>>4.
// elems 0..7  = K g*8..g*8+7 ; elems 8..15 = K 16+g*8..+7  (base = row + kk*32)
static __device__ inline v16h lds_a_frag(const _Float16* base, int g) {
  v8h x0 = *(const v8h*)(base + g * 8);
  v8h x1 = *(const v8h*)(base + 16 + g * 8);
  return combine16(x0, x1);
}
// B fragment (32x16 f16): lane -> col n = lane&15, g = lane>>4 holds
// elems 0..7 = K g*8..g*8+7 ; elems 8..15 = K 16+g*8..+7 (base = row n + kk*32)
static __device__ inline v16h lds_b_frag(const _Float16* base, int g) {
  v8h x0 = *(const v8h*)(base + g * 8);
  v8h x1 = *(const v8h*)(base + 16 + g * 8);
  return combine16(x0, x1);
}

// ---------------------------------------------------------------------------
// Stage 4: WMMA GEMM  Y[m][n] = sum_k X[m][k] * W[n][k]
// 256 threads = 8 waves; block tile 128Mx64N; K staged 64 at a time into LDS
// via async copies, double-buffered. Each wave owns a 16x64 strip (4 acc
// tiles), 3-pass hi/lo per k-sub: ah*bh + ah*bl + al*bh.
// ---------------------------------------------------------------------------
__global__ __launch_bounds__(256) void ortho_wmma_gemm(
    const _Float16* __restrict__ Xhi, const _Float16* __restrict__ Xlo,
    const _Float16* __restrict__ Whi, const _Float16* __restrict__ Wlo,
    float* __restrict__ Y) {
  __shared__ __align__(16) _Float16 sAhi[2][BM * LDA];
  __shared__ __align__(16) _Float16 sAlo[2][BM * LDA];
  __shared__ __align__(16) _Float16 sBhi[2][BN * LDA];
  __shared__ __align__(16) _Float16 sBlo[2][BN * LDA];

  const int t    = threadIdx.x;
  const int lane = t & 31;
  const int wave = t >> 5;
  const int m16  = lane & 15;
  const int g    = lane >> 4;

  const int mBlock = blockIdx.y * BM;
  const int nBlock = blockIdx.x * BN;

  v8f acc[4] = {v8f{}, v8f{}, v8f{}, v8f{}};

  // ---- staging lambda: copy one KSTEP slab of A(hi/lo) + B(hi/lo) ----
  auto stage = [&](int buf, int kb) {
    const int k0 = kb * KSTEP;
    // A: BM rows x KSTEP f16 = 1024 chunks of 8 f16; 4 per thread, coalesced
#pragma unroll
    for (int i = 0; i < 4; ++i) {
      int c = i * 256 + t;
      int row = c >> 3, cx = c & 7;
      size_t goff = (size_t)(mBlock + row) * IN_F + k0 + cx * 8;
      int loff = row * LDA + cx * 8;
      cp16_async(Xhi + goff, &sAhi[buf][loff]);
      cp16_async(Xlo + goff, &sAlo[buf][loff]);
    }
    // B: BN rows x KSTEP f16 = 512 chunks; 2 per thread
#pragma unroll
    for (int i = 0; i < 2; ++i) {
      int c = i * 256 + t;
      int row = c >> 3, cx = c & 7;
      size_t goff = (size_t)(nBlock + row) * IN_F + k0 + cx * 8;
      int loff = row * LDA + cx * 8;
      cp16_async(Whi + goff, &sBhi[buf][loff]);
      cp16_async(Wlo + goff, &sBlo[buf][loff]);
    }
  };

  stage(0, 0);  // prologue

  const int KSTEPS = IN_F / KSTEP;  // 64
  for (int kb = 0; kb < KSTEPS; ++kb) {
    const int cur = kb & 1;
    wait_async_all();   // own async writes landed in LDS
    __syncthreads();    // everyone's staging done; prev buffer fully consumed
    if (kb + 1 < KSTEPS) stage(cur ^ 1, kb + 1);  // overlap next slab

    const _Float16* aHiRow = &sAhi[cur][(wave * 16 + m16) * LDA];
    const _Float16* aLoRow = &sAlo[cur][(wave * 16 + m16) * LDA];
#pragma unroll
    for (int kk = 0; kk < 2; ++kk) {
      v16h ah = lds_a_frag(aHiRow + kk * 32, g);
      v16h al = lds_a_frag(aLoRow + kk * 32, g);
#pragma unroll
      for (int nt = 0; nt < 4; ++nt) {
        const int brow = (nt * 16 + m16) * LDA + kk * 32;
        v16h bh = lds_b_frag(&sBhi[cur][brow], g);
        v16h bl = lds_b_frag(&sBlo[cur][brow], g);
        acc[nt] = __builtin_amdgcn_wmma_f32_16x16x32_f16(
            false, ah, false, bh, (short)0, acc[nt], false, false);
        acc[nt] = __builtin_amdgcn_wmma_f32_16x16x32_f16(
            false, ah, false, bl, (short)0, acc[nt], false, false);
        acc[nt] = __builtin_amdgcn_wmma_f32_16x16x32_f16(
            false, al, false, bh, (short)0, acc[nt], false, false);
      }
    }
  }

  // C/D layout: VGPR j -> row = g*8 + j, col = lane&15
  const int mBase = mBlock + wave * 16;
#pragma unroll
  for (int nt = 0; nt < 4; ++nt) {
    float* yp = Y + (size_t)(mBase + g * 8) * OUT_F + nBlock + nt * 16 + m16;
#pragma unroll
    for (int j = 0; j < 8; ++j) yp[(size_t)j * OUT_F] = acc[nt][j];
  }
}

// ---------------------------------------------------------------------------
// Launch. Workspace (128 MB):
//   [0,64MB)   : W f32 (stages 1-3)  -> reused as Xhi/Xlo afterwards
//   [64,96MB)  : W hi f16
//   [96,128MB) : W lo f16
// ---------------------------------------------------------------------------
extern "C" void kernel_launch(void* const* d_in, const int* in_sizes, int n_in,
                              void* d_out, int out_size, void* d_ws, size_t ws_size,
                              hipStream_t stream) {
  const float* x      = (const float*)d_in[0];
  const int*   packed = (const int*)d_in[1];
  const float* scales = (const float*)d_in[2];
  const float* vals   = (const float*)d_in[3];
  const int*   rows   = (const int*)d_in[4];
  const int*   cols   = (const int*)d_in[5];
  float* y = (float*)d_out;
  const int nnz = in_sizes[3];

  char* ws = (char*)d_ws;
  float*    Wf32 = (float*)ws;
  _Float16* Whi  = (_Float16*)(ws + ((size_t)64 << 20));
  _Float16* Wlo  = (_Float16*)(ws + ((size_t)96 << 20));
  _Float16* Xhi  = (_Float16*)ws;                         // aliases Wf32 (done)
  _Float16* Xlo  = (_Float16*)(ws + ((size_t)32 << 20));

  ortho_dequant<<<(OUT_F * (IN_F / 2) + 255) / 256, 256, 0, stream>>>(packed, scales, Wf32);
  ortho_scatter<<<(nnz + 255) / 256, 256, 0, stream>>>(vals, rows, cols, Wf32, nnz);
  ortho_split<<<(OUT_F * IN_F + 255) / 256, 256, 0, stream>>>(Wf32, Whi, Wlo, OUT_F * IN_F);
  ortho_split<<<(TOKENS * IN_F + 255) / 256, 256, 0, stream>>>(x, Xhi, Xlo, TOKENS * IN_F);

  dim3 grid(OUT_F / BN, TOKENS / BM);
  ortho_wmma_gemm<<<grid, 256, 0, stream>>>(Xhi, Xlo, Whi, Wlo, y);
}
